// TriangularMultiplicativeUpdate_21741124452577
// MI455X (gfx1250) — matrix operand order repaired and checked by reading, not compiled, not run on hardware
//
#include <hip/hip_runtime.h>

// ---------------------------------------------------------------------------
// Triangular multiplicative update (outgoing) for MI455X / gfx1250.
// All contractions run on v_wmma_f32_16x16x32_bf16 (wave32 WMMA).
// The batched-einsum kernel double-buffers its LDS tiles with CDNA5
// GLOBAL_LOAD_ASYNC_TO_LDS_B128 (ASYNCcnt) when the builtins are available.
// N = 512, C = c_z = c_h = 128.
// ---------------------------------------------------------------------------

typedef __bf16 bf16_t;
typedef __attribute__((ext_vector_type(16))) __bf16 v16bf;
typedef __attribute__((ext_vector_type(8)))  __bf16 v8bf;   // 16 bytes
typedef __attribute__((ext_vector_type(4)))  __bf16 v4bf;   // 8 bytes
typedef __attribute__((ext_vector_type(8)))  float  v8f;
typedef __attribute__((ext_vector_type(4)))  float  v4f;
typedef __attribute__((ext_vector_type(4)))  int    v4i;    // 16 bytes

#define NPT   512
#define CCH   128
#define NNTOT (512*512)

// ---- CDNA5 async global->LDS path (guarded; falls back to sync staging) ----
#if defined(__HIP_DEVICE_COMPILE__)
#  if __has_builtin(__builtin_amdgcn_global_load_async_to_lds_b128) && \
      __has_builtin(__builtin_amdgcn_s_wait_asynccnt)
#    define ASYNC_LDS_OK 1
#  endif
#endif

#ifdef ASYNC_LDS_OK
// Builtin signature (from the toolchain's diagnostic): the global address is a
// pointer to int4 in address space 1, the LDS address a pointer to int4 in
// address space 3 (generic->AS1 keeps the full VA; generic->AS3 keeps the low
// 32 bits = wave-relative LDS byte address per the flat-aperture rules).
__device__ __forceinline__ void async_cp16(const void* g, void* l) {
  __attribute__((address_space(1))) v4i* gp =
      (__attribute__((address_space(1))) v4i*)(unsigned long long)(size_t)g;
  __attribute__((address_space(3))) v4i* lp =
      (__attribute__((address_space(3))) v4i*)(unsigned int)(size_t)l;
  __builtin_amdgcn_global_load_async_to_lds_b128(gp, lp, 0, 0);
}
#define WAIT_ASYNC(n) __builtin_amdgcn_s_wait_asynccnt(n)
#else
#define WAIT_ASYNC(n)
#endif

// Copy 32 bytes of L-row and 32 bytes of R-row for this thread into LDS.
__device__ __forceinline__ void stage_tile(const bf16_t* gl, const bf16_t* gr,
                                           bf16_t* ll, bf16_t* lr) {
#ifdef ASYNC_LDS_OK
  async_cp16(gl,     ll);
  async_cp16(gl + 8, ll + 8);
  async_cp16(gr,     lr);
  async_cp16(gr + 8, lr + 8);
#else
  *(v8bf*)ll       = *(const v8bf*)gl;
  *(v8bf*)(ll + 8) = *(const v8bf*)(gl + 8);
  *(v8bf*)lr       = *(const v8bf*)gr;
  *(v8bf*)(lr + 8) = *(const v8bf*)(gr + 8);
#endif
}

// ---- WMMA wrapper ----------------------------------------------------------
__device__ __forceinline__ v8f wmma_bf16(v16bf a, v16bf b, v8f c) {
  return __builtin_amdgcn_wmma_f32_16x16x32_bf16(
      /*neg_a=*/false, a, /*neg_b=*/false, b,
      /*c_mod=*/(short)0, c, /*reuse_a=*/false, /*reuse_b=*/false);
}

// A-fragment (16x32 bf16, M rows striped over lanes):
//   lanes 0-15 : row = lane,     K = k0+0..7  (v0-3), k0+16..23 (v4-7)
//   lanes16-31 : row = lane-16,  K = k0+8..15 (v0-3), k0+24..31 (v4-7)
__device__ __forceinline__ v16bf load_a(const bf16_t* base, int ld, int k0, int lane) {
  int row = lane & 15;
  int ko  = k0 + ((lane >> 4) << 3);
  const bf16_t* q = base + row * ld + ko;
  v8bf lo = *(const v8bf*)(q);
  v8bf hi = *(const v8bf*)(q + 16);
  v16bf r;
#pragma unroll
  for (int i = 0; i < 8; ++i) { r[i] = lo[i]; r[i + 8] = hi[i]; }
  return r;
}

// Same fragment but sourced from fp32 memory (converted on the fly).
__device__ __forceinline__ v16bf load_a_f32(const float* base, int ld, int k0, int lane) {
  int row = lane & 15;
  int ko  = k0 + ((lane >> 4) << 3);
  const float* q = base + row * ld + ko;
  v4f a0 = *(const v4f*)(q);
  v4f a1 = *(const v4f*)(q + 4);
  v4f a2 = *(const v4f*)(q + 16);
  v4f a3 = *(const v4f*)(q + 20);
  v16bf r;
#pragma unroll
  for (int e = 0; e < 4; ++e) {
    r[e]      = (bf16_t)a0[e];
    r[4 + e]  = (bf16_t)a1[e];
    r[8 + e]  = (bf16_t)a2[e];
    r[12 + e] = (bf16_t)a3[e];
  }
  return r;
}

// B-fragment (32x16 bf16, N cols striped over lanes) loaded from "B^T"
// storage, i.e. memory rows are the N dimension with K contiguous:
//   lanes 0-15 : col = lane,     K = k0+0..7  (v0-3), k0+16..23 (v4-7)
//   lanes16-31 : col = lane-16,  K = k0+8..15 (v0-3), k0+24..31 (v4-7)
__device__ __forceinline__ v16bf load_b(const bf16_t* base, int ld, int k0, int lane) {
  int col = lane & 15;
  int ko  = k0 + ((lane >> 4) << 3);
  const bf16_t* q = base + col * ld + ko;
  v8bf lo = *(const v8bf*)(q);
  v8bf hi = *(const v8bf*)(q + 16);
  v16bf r;
#pragma unroll
  for (int i = 0; i < 8; ++i) { r[i] = lo[i]; r[i + 8] = hi[i]; }
  return r;
}

__device__ __forceinline__ float sigmoidf_(float x) {
  return 1.0f / (1.0f + __expf(-x));
}

// ---------------------------------------------------------------------------
// K0: convert 6 weight matrices (128x128 fp32, [k][n]) to bf16 transposed
// [n][k] so WMMA B-fragments load contiguously.  Order: lp, lg, rp, rg, og, op
// ---------------------------------------------------------------------------
__global__ void wconv(const float* __restrict__ w0, const float* __restrict__ w1,
                      const float* __restrict__ w2, const float* __restrict__ w3,
                      const float* __restrict__ w4, const float* __restrict__ w5,
                      bf16_t* __restrict__ wt) {
  const float* srcs[6] = {w0, w1, w2, w3, w4, w5};
  int mat = blockIdx.x >> 7;   // 0..5
  int n   = blockIdx.x & 127;
  int k   = threadIdx.x;       // 0..127
  wt[mat * 16384 + n * 128 + k] = (bf16_t)srcs[mat][k * 128 + n];
}

// ---------------------------------------------------------------------------
// K1: z = bf16(layer_norm(pair))   row-major [r][c], r = i*512+j
// ---------------------------------------------------------------------------
__global__ __launch_bounds__(256) void lnz(const float* __restrict__ pair,
                                           const float* __restrict__ g,
                                           const float* __restrict__ b,
                                           bf16_t* __restrict__ z) {
  int tid = threadIdx.x, wave = tid >> 5, lane = tid & 31;
  size_t r0 = (size_t)blockIdx.x * 128;
  v4f gv = *(const v4f*)(g + lane * 4);
  v4f bv = *(const v4f*)(b + lane * 4);
#pragma unroll 4
  for (int i = 0; i < 16; ++i) {
    size_t r = r0 + wave * 16 + i;
    v4f x = *(const v4f*)(pair + r * CCH + lane * 4);
    float s  = x[0] + x[1] + x[2] + x[3];
    float s2 = x[0]*x[0] + x[1]*x[1] + x[2]*x[2] + x[3]*x[3];
#pragma unroll
    for (int o = 16; o > 0; o >>= 1) {
      s  += __shfl_xor(s,  o, 32);
      s2 += __shfl_xor(s2, o, 32);
    }
    float mu = s * (1.0f / 128.0f);
    float rs = rsqrtf(s2 * (1.0f / 128.0f) - mu * mu + 1e-5f);
    v4bf zz;
#pragma unroll
    for (int j = 0; j < 4; ++j)
      zz[j] = (bf16_t)((x[j] - mu) * rs * gv[j] + bv[j]);
    *(v4bf*)(z + r * CCH + lane * 4) = zz;
  }
}

// ---------------------------------------------------------------------------
// K2: left  = (z@lp_w + lp_b) * sigmoid(z@lg_w + lg_b)
//     right = (z@rp_w + rp_b) * sigmoid(z@rg_w + rg_b)
// Stored CHANNEL-MAJOR: left_t[c*NN + r].  Block = 128 rows, 8 waves; each
// wave owns 16 rows x full 128-channel output.  Result transposed through LDS
// for coalesced channel-major stores.
// ---------------------------------------------------------------------------
__global__ __launch_bounds__(256) void proj(const bf16_t* __restrict__ z,
                                            const bf16_t* __restrict__ wt,
                                            const float* __restrict__ lp_b,
                                            const float* __restrict__ lg_b,
                                            const float* __restrict__ rp_b,
                                            const float* __restrict__ rg_b,
                                            bf16_t* __restrict__ left_t,
                                            bf16_t* __restrict__ right_t) {
  __shared__ bf16_t ob[128][132];             // 33792 B
  int tid = threadIdx.x, wave = tid >> 5, lane = tid & 31;
  size_t r0 = (size_t)blockIdx.x * 128;
  int m0 = wave * 16;
  const bf16_t* zrow = z + (r0 + m0) * CCH;
  int hl = lane >> 4, cn = lane & 15;
  int cc = tid >> 1, half = tid & 1;

  for (int side = 0; side < 2; ++side) {
    const bf16_t* wp = wt + (side * 2 + 0) * 16384;
    const bf16_t* wg = wt + (side * 2 + 1) * 16384;
    const float* bp = side ? rp_b : lp_b;
    const float* bg = side ? rg_b : lg_b;
    v8f accp[8], accg[8];
#pragma unroll
    for (int nt = 0; nt < 8; ++nt)
#pragma unroll
      for (int e = 0; e < 8; ++e) { accp[nt][e] = 0.0f; accg[nt][e] = 0.0f; }

    for (int k0 = 0; k0 < 128; k0 += 32) {
      v16bf a = load_a(zrow, CCH, k0, lane);
#pragma unroll
      for (int nt = 0; nt < 8; ++nt) {
        v16bf bpf = load_b(wp + nt * 2048, 128, k0, lane);
        v16bf bgf = load_b(wg + nt * 2048, 128, k0, lane);
        accp[nt] = wmma_bf16(a, bpf, accp[nt]);
        accg[nt] = wmma_bf16(a, bgf, accg[nt]);
      }
    }

    __syncthreads();   // side 1: previous LDS readers are done
#pragma unroll
    for (int nt = 0; nt < 8; ++nt) {
      int col = nt * 16 + cn;
      float bpc = bp[col], bgc = bg[col];
#pragma unroll
      for (int v = 0; v < 8; ++v) {
        int row = m0 + v + hl * 8;
        float pv = accp[nt][v] + bpc;
        float gv = accg[nt][v] + bgc;
        ob[row][col] = (bf16_t)(pv * sigmoidf_(gv));
      }
    }
    __syncthreads();

    // channel-major store: thread (c, half) writes 64 consecutive r
    bf16_t* dst = (side ? right_t : left_t) + (size_t)cc * NNTOT + r0 + half * 64;
#pragma unroll
    for (int blk = 0; blk < 8; ++blk) {
      v8bf pk;
#pragma unroll
      for (int e = 0; e < 8; ++e) pk[e] = ob[half * 64 + blk * 8 + e][cc];
      *(v8bf*)(dst + blk * 8) = pk;
    }
  }
}

// ---------------------------------------------------------------------------
// K3: batched einsum  out[c][i][j] = sum_k L_c[i][k] * R_c[j][k]
// 128 channels x 16 (128x128) tiles = 2048 blocks.  8 waves, each a 32x64
// sub-tile (2x4 WMMA tiles).  k-loop of 32 through DOUBLE-BUFFERED LDS fed by
// async global->LDS copies (s_wait_asynccnt 4 keeps next tile in flight).
// ---------------------------------------------------------------------------
__global__ __launch_bounds__(256) void tri(const bf16_t* __restrict__ left_t,
                                           const bf16_t* __restrict__ right_t,
                                           float* __restrict__ out_t) {
  __shared__ bf16_t Lt[2][128][40];           // 20480 B
  __shared__ bf16_t Rt[2][128][40];           // 20480 B
  int tid = threadIdx.x, wave = tid >> 5, lane = tid & 31;
  int cch  = blockIdx.x >> 4;
  int tile = blockIdx.x & 15;
  int i0 = (tile >> 2) << 7;
  int j0 = (tile & 3) << 7;
  const bf16_t* L = left_t  + (size_t)cch * NNTOT;
  const bf16_t* R = right_t + (size_t)cch * NNTOT;
  int wi = wave >> 1, wj = wave & 1;

  v8f acc[2][4];
#pragma unroll
  for (int mt = 0; mt < 2; ++mt)
#pragma unroll
    for (int nt = 0; nt < 4; ++nt)
#pragma unroll
      for (int e = 0; e < 8; ++e) acc[mt][nt][e] = 0.0f;

  int lrow = tid >> 1, lp2 = (tid & 1) * 16;
  const bf16_t* gl = L + (size_t)(i0 + lrow) * NPT + lp2;
  const bf16_t* gr = R + (size_t)(j0 + lrow) * NPT + lp2;

  // prologue: stage k-step 0 into buffer 0
  stage_tile(gl, gr, &Lt[0][lrow][lp2], &Rt[0][lrow][lp2]);

  for (int ks = 0; ks < 16; ++ks) {
    int cur = ks & 1;
    if (ks < 15) {
      stage_tile(gl + (ks + 1) * 32, gr + (ks + 1) * 32,
                 &Lt[cur ^ 1][lrow][lp2], &Rt[cur ^ 1][lrow][lp2]);
      WAIT_ASYNC(4);     // wait for current buffer; next buffer stays in flight
    } else {
      WAIT_ASYNC(0);     // last step: drain everything
    }
    __syncthreads();

    v16bf af0 = load_a(&Lt[cur][wi * 32][0],      40, 0, lane);
    v16bf af1 = load_a(&Lt[cur][wi * 32 + 16][0], 40, 0, lane);
#pragma unroll
    for (int nt = 0; nt < 4; ++nt) {
      v16bf bfv = load_b(&Rt[cur][wj * 64 + nt * 16][0], 40, 0, lane);
      acc[0][nt] = wmma_bf16(af0, bfv, acc[0][nt]);
      acc[1][nt] = wmma_bf16(af1, bfv, acc[1][nt]);
    }
    __syncthreads();     // readers done before this buffer is re-staged
  }

  float* O = out_t + (size_t)cch * NNTOT;
  int hl = lane >> 4, cn = lane & 15;
#pragma unroll
  for (int mt = 0; mt < 2; ++mt)
#pragma unroll
    for (int nt = 0; nt < 4; ++nt)
#pragma unroll
      for (int v = 0; v < 8; ++v) {
        int row = i0 + wi * 32 + mt * 16 + v + hl * 8;
        int col = j0 + wj * 64 + nt * 16 + cn;
        O[(size_t)row * NPT + col] = acc[mt][nt][v];
      }
}

// ---------------------------------------------------------------------------
// K4: transpose out_t [c][r] (fp32) -> out_rm [r][c] (fp32) via 128x32 tiles
// ---------------------------------------------------------------------------
__global__ __launch_bounds__(256) void transk(const float* __restrict__ out_t,
                                              float* __restrict__ out_rm) {
  __shared__ float tile[128][33];             // 16896 B
  int bx = blockIdx.x;                        // 8192 = 2048 r-blocks * 4 c-blocks
  int cb = bx & 3, rb = bx >> 2;
  size_t r0 = (size_t)rb * 128;
  int c0 = cb * 32;
  int t = threadIdx.x;
  {
    int cl = t >> 3, rp = t & 7;              // 32 cols, 16 rows each
    const float* src = out_t + (size_t)(c0 + cl) * NNTOT + r0 + rp * 16;
#pragma unroll
    for (int q = 0; q < 4; ++q) {
      v4f v = *(const v4f*)(src + q * 4);
#pragma unroll
      for (int e = 0; e < 4; ++e) tile[rp * 16 + q * 4 + e][cl] = v[e];
    }
  }
  __syncthreads();
  {
    int rl = t >> 1, ch = t & 1;              // 128 rows, 16 cols each
    float* dst = out_rm + (r0 + rl) * CCH + c0 + ch * 16;
#pragma unroll
    for (int q = 0; q < 4; ++q) {
      v4f w;
#pragma unroll
      for (int e = 0; e < 4; ++e) w[e] = tile[rl][ch * 16 + q * 4 + e];
      *(v4f*)(dst + q * 4) = w;
    }
  }
}

// ---------------------------------------------------------------------------
// K5: final fused stage:
//   h    = layer_norm(out_rm, fn_g, fn_b)
//   o    = h @ op_w + op_b            (WMMA, A from LDS bf16)
//   gate = sigmoid(pair @ og_w + og_b)(WMMA, A fp32->bf16 in regs)
//   out  = pair + gate * o
// ---------------------------------------------------------------------------
__global__ __launch_bounds__(256) void fin(const float* __restrict__ out_rm,
                                           const float* __restrict__ pair,
                                           const float* __restrict__ fg,
                                           const float* __restrict__ fb,
                                           const bf16_t* __restrict__ wt_og,
                                           const bf16_t* __restrict__ wt_op,
                                           const float* __restrict__ ogb,
                                           const float* __restrict__ opb,
                                           float* __restrict__ out) {
  __shared__ bf16_t zb[128][136];             // 34816 B
  int tid = threadIdx.x, wave = tid >> 5, lane = tid & 31;
  size_t r0 = (size_t)blockIdx.x * 128;

  v4f gv = *(const v4f*)(fg + lane * 4);
  v4f bv = *(const v4f*)(fb + lane * 4);
#pragma unroll 4
  for (int i = 0; i < 16; ++i) {
    int row = wave * 16 + i;
    v4f x = *(const v4f*)(out_rm + (r0 + row) * CCH + lane * 4);
    float s  = x[0] + x[1] + x[2] + x[3];
    float s2 = x[0]*x[0] + x[1]*x[1] + x[2]*x[2] + x[3]*x[3];
#pragma unroll
    for (int o = 16; o > 0; o >>= 1) {
      s  += __shfl_xor(s,  o, 32);
      s2 += __shfl_xor(s2, o, 32);
    }
    float mu = s * (1.0f / 128.0f);
    float rs = rsqrtf(s2 * (1.0f / 128.0f) - mu * mu + 1e-5f);
#pragma unroll
    for (int j = 0; j < 4; ++j)
      zb[row][lane * 4 + j] = (bf16_t)((x[j] - mu) * rs * gv[j] + bv[j]);
  }
  __syncthreads();

  int m0 = wave * 16;
  const float* prow = pair + (r0 + m0) * CCH;
  v8f acco[8], accg[8];
#pragma unroll
  for (int nt = 0; nt < 8; ++nt)
#pragma unroll
    for (int e = 0; e < 8; ++e) { acco[nt][e] = 0.0f; accg[nt][e] = 0.0f; }

  for (int k0 = 0; k0 < 128; k0 += 32) {
    v16bf az = load_a(&zb[m0][0], 136, k0, lane);
    v16bf ap = load_a_f32(prow, CCH, k0, lane);
#pragma unroll
    for (int nt = 0; nt < 8; ++nt) {
      acco[nt] = wmma_bf16(az, load_b(wt_op + nt * 2048, 128, k0, lane), acco[nt]);
      accg[nt] = wmma_bf16(ap, load_b(wt_og + nt * 2048, 128, k0, lane), accg[nt]);
    }
  }

  int hl = lane >> 4, cn = lane & 15;
#pragma unroll
  for (int nt = 0; nt < 8; ++nt) {
    int col = nt * 16 + cn;
    float bo = opb[col], bg2 = ogb[col];
#pragma unroll
    for (int v = 0; v < 8; ++v) {
      int row = m0 + v + hl * 8;
      size_t idx = (r0 + row) * CCH + col;
      float gate = sigmoidf_(accg[nt][v] + bg2);
      out[idx] = pair[idx] + gate * (acco[nt][v] + bo);
    }
  }
}

// ---------------------------------------------------------------------------
// Host launcher.  Workspace layout (bytes):
//   [0, 196608)       : wt   (6 x 128x128 bf16, transposed)
//   [196608, +64MB)   : z    bf16   (aliased later by out_rm)
//   [+64MB, +128MB)   : left_t  bf16 (aliased later by out_rm)
//   [+128MB, +192MB)  : right_t bf16
//   [+192MB, +320MB)  : out_t   fp32 (channel-major einsum out)
//   out_rm (fp32, 128MB) aliases [196608, +128MB) -- z/left_t dead by then.
// ---------------------------------------------------------------------------
extern "C" void kernel_launch(void* const* d_in, const int* in_sizes, int n_in,
                              void* d_out, int out_size, void* d_ws, size_t ws_size,
                              hipStream_t stream) {
  const float* pair = (const float*)d_in[0];
  const float* ln_g = (const float*)d_in[1];
  const float* ln_b = (const float*)d_in[2];
  const float* lp_w = (const float*)d_in[3];
  const float* lp_b = (const float*)d_in[4];
  const float* rp_w = (const float*)d_in[5];
  const float* rp_b = (const float*)d_in[6];
  const float* lg_w = (const float*)d_in[7];
  const float* lg_b = (const float*)d_in[8];
  const float* rg_w = (const float*)d_in[9];
  const float* rg_b = (const float*)d_in[10];
  const float* og_w = (const float*)d_in[11];
  const float* og_b = (const float*)d_in[12];
  const float* op_w = (const float*)d_in[13];
  const float* op_b = (const float*)d_in[14];
  const float* fn_g = (const float*)d_in[15];
  const float* fn_b = (const float*)d_in[16];

  char* ws = (char*)d_ws;
  const size_t MB64 = 67108864ull;
  bf16_t* wt      = (bf16_t*)ws;
  bf16_t* zbuf    = (bf16_t*)(ws + 196608);
  bf16_t* left_t  = (bf16_t*)(ws + 196608 + 1 * MB64);
  bf16_t* right_t = (bf16_t*)(ws + 196608 + 2 * MB64);
  float*  out_t   = (float*) (ws + 196608 + 3 * MB64);
  float*  out_rm  = (float*) (ws + 196608);          // aliases zbuf+left_t
  float*  outp    = (float*)d_out;

  wconv <<<dim3(6 * 128), dim3(128), 0, stream>>>(lp_w, lg_w, rp_w, rg_w, og_w, op_w, wt);
  lnz   <<<dim3(2048),    dim3(256), 0, stream>>>(pair, ln_g, ln_b, zbuf);
  proj  <<<dim3(2048),    dim3(256), 0, stream>>>(zbuf, wt, lp_b, lg_b, rp_b, rg_b, left_t, right_t);
  tri   <<<dim3(2048),    dim3(256), 0, stream>>>(left_t, right_t, out_t);
  transk<<<dim3(8192),    dim3(256), 0, stream>>>(out_t, out_rm);
  fin   <<<dim3(2048),    dim3(256), 0, stream>>>(out_rm, pair, fn_g, fn_b,
                                                  wt + 4 * 16384, wt + 5 * 16384,
                                                  og_b, op_b, outp);
}
